// DIN_41214506172784
// MI455X (gfx1250) — matrix operand order repaired and checked by reading, not compiled, not run on hardware
//
#include <hip/hip_runtime.h>

#define B_TOTAL 16384
#define S_SEQ 30
#define SP 32
#define BT 8
#define THREADS 128

typedef __attribute__((ext_vector_type(16))) __bf16 v16bf;
typedef __attribute__((ext_vector_type(2)))  __bf16 v2bf;
typedef __attribute__((ext_vector_type(8)))  float  v8f;

// Native bf16 converts (RTE) — let ISel pick v_cvt_pk_bf16_f32 where available.
__device__ __forceinline__ unsigned short f2bf(float f) {
  return __builtin_bit_cast(unsigned short, (__bf16)f);
}
__device__ __forceinline__ unsigned int packbf2(float v0, float v1) {
  v2bf p;
  p[0] = (__bf16)v0;
  p[1] = (__bf16)v1;
  return __builtin_bit_cast(unsigned int, p);
}
__device__ __forceinline__ float bf2f(unsigned short b) {
  unsigned int u = ((unsigned int)b) << 16;
  return __builtin_bit_cast(float, u);
}

__device__ __forceinline__ v8f wmma_bf16(v16bf a, v16bf b, v8f c) {
  return __builtin_amdgcn_wmma_f32_16x16x32_bf16(false, a, false, b, (short)0, c,
                                                 false, false);
}

// ---- B-fragment swizzle: W[K][N] fp32 -> bf16 fragments in WMMA B layout.
// B matrix 32x16: lane 0-15 -> N, VGPR 0-3: K 0..7 (lanes16-31: 8..15), VGPR 4-7: +16.
// Fragment (kt,nt) = 256 dwords: dword index = lane*8 + j.
__global__ void DIN_swizzleW(const float* __restrict__ W, unsigned int* __restrict__ dst,
                             int K, int N) {
  int tid = blockIdx.x * blockDim.x + threadIdx.x;
  int NT = N >> 4;
  int total = (K >> 5) * NT * 256;
  if (tid >= total) return;
  int frag = tid >> 8;
  int r    = tid & 255;
  int lane = r >> 3;
  int j    = r & 7;
  int kt = frag / NT, nt = frag % NT;
  int k = kt * 32 + ((lane >= 16) ? 8 : 0) + ((j >= 4) ? 16 : 0) + 2 * (j & 3);
  int n = nt * 16 + (lane & 15);
  dst[tid] = packbf2(W[(size_t)k * N + n], W[(size_t)(k + 1) * N + n]);
}

__device__ __forceinline__ v16bf load_b_frag(const unsigned int* base, int lane) {
  union { v16bf v; uint4 q[2]; } f;
  const uint4* p = (const uint4*)(base + lane * 8);
  f.q[0] = p[0];
  f.q[1] = p[1];
  return f.v;
}

// A matrix 16x32: lane 0-15 -> M, VGPR 0-3: K 0..7 (lanes16-31: 8..15), VGPR 4-7: +16.
__device__ __forceinline__ v16bf build_a_frag(const unsigned short* mat, int ncols,
                                              int kt, int lane) {
  union { v16bf v; unsigned int u[8]; } f;
  int kbase = kt * 32 + ((lane >= 16) ? 8 : 0);
  const unsigned short* row = mat + (lane & 15) * ncols;
#pragma unroll
  for (int j = 0; j < 8; ++j) {
    int c0 = kbase + ((j >= 4) ? 16 : 0) + 2 * (j & 3);
    f.u[j] = *(const unsigned int*)(row + c0);
  }
  return f.v;
}

// Attention feature A-fragment: feats = [q | k | q*k | q-k] synthesized per column.
__device__ __forceinline__ v16bf build_feat_frag(const unsigned short* qrow,
                                                 const unsigned short* krow,
                                                 int kt, int lane) {
  union { v16bf v; unsigned int u[8]; } f;
  int kbase = kt * 32 + ((lane >= 16) ? 8 : 0);
#pragma unroll
  for (int j = 0; j < 8; ++j) {
    int c0 = kbase + ((j >= 4) ? 16 : 0) + 2 * (j & 3);
    int reg = c0 >> 6;
    int d   = c0 & 63;
    if (reg == 0) {               // query columns already bf16: copy packed pair
      f.u[j] = *(const unsigned int*)(qrow + d);
    } else if (reg == 1) {        // key columns already bf16
      f.u[j] = *(const unsigned int*)(krow + d);
    } else {
      float q0 = bf2f(qrow[d]),     k0 = bf2f(krow[d]);
      float q1 = bf2f(qrow[d + 1]), k1 = bf2f(krow[d + 1]);
      f.u[j] = (reg == 2) ? packbf2(q0 * k0, q1 * k1)
                          : packbf2(q0 - k0, q1 - k1);
    }
  }
  return f.v;
}

__global__ __launch_bounds__(THREADS)
void DIN_fused(const int* __restrict__ user_ids, const int* __restrict__ target_items,
               const int* __restrict__ target_cates, const int* __restrict__ seq_items,
               const int* __restrict__ seq_cates, const int* __restrict__ seq_behaviors,
               const int* __restrict__ seq_lengths,
               const float* __restrict__ user_emb, const float* __restrict__ item_emb,
               const float* __restrict__ cate_emb, const float* __restrict__ behavior_emb,
               const float* __restrict__ ab0, const float* __restrict__ ab1,
               const float* __restrict__ aW2, const float* __restrict__ ab2,
               const float* __restrict__ mb0, const float* __restrict__ mb1,
               const float* __restrict__ mW2, const float* __restrict__ mb2,
               const unsigned int* __restrict__ wsA0, const unsigned int* __restrict__ wsA1,
               const unsigned int* __restrict__ wsM0, const unsigned int* __restrict__ wsM1,
               float* __restrict__ out, float* __restrict__ attn_out) {
  // LDS carve-up (59392 B total)
  __shared__ __align__(16) unsigned char smem[59392];
  unsigned short* keysb  = (unsigned short*)smem;                 // [8][32][64] bf16
  unsigned short* qb     = (unsigned short*)(smem + 32768);       // [8][64] bf16
  unsigned short* concat = (unsigned short*)(smem + 33792);       // [16][256] bf16
  float*          attn   = (float*)(smem + 41984);                // [8][32] f32
  unsigned short* hz     = (unsigned short*)(smem + 43008);       // 16384 B scratch

  const int tid  = threadIdx.x;
  const int lane = tid & 31;
  const int wave = tid >> 5;
  const int b0   = blockIdx.x * BT;

  // ---------------- Phase 0: gathers into LDS (bf16) ----------------
  for (int row = tid; row < BT * SP; row += THREADS) {
    int bl = row >> 5, s = row & 31;
    unsigned int* dst = (unsigned int*)(keysb + row * 64);
    if (s < S_SEQ) {
      int b = b0 + bl;
      int it = seq_items[b * S_SEQ + s];
      int ct = seq_cates[b * S_SEQ + s];
      int bh = seq_behaviors[b * S_SEQ + s];
      const float4* ip = (const float4*)(item_emb + (size_t)it * 64);
      const float4* cp = (const float4*)(cate_emb + (size_t)ct * 64);
      const float4* bp = (const float4*)(behavior_emb + (size_t)bh * 64);
#pragma unroll 4
      for (int q = 0; q < 16; ++q) {
        float4 a = ip[q], c = cp[q], e = bp[q];
        dst[q * 2 + 0] = packbf2(a.x + c.x + e.x, a.y + c.y + e.y);
        dst[q * 2 + 1] = packbf2(a.z + c.z + e.z, a.w + c.w + e.w);
      }
    } else {
#pragma unroll
      for (int q = 0; q < 32; ++q) dst[q] = 0u;   // zero padded seq rows
    }
  }
  for (int i = tid; i < BT * 64; i += THREADS) {   // user/tgt gathers -> concat + query
    int bl = i >> 6, d = i & 63;
    int b = b0 + bl;
    float uv = user_emb[(size_t)user_ids[b] * 64 + d];
    float ti = item_emb[(size_t)target_items[b] * 64 + d];
    float tc = cate_emb[(size_t)target_cates[b] * 64 + d];
    unsigned short tib = f2bf(ti);
    concat[bl * 256 + d]       = f2bf(uv);
    concat[bl * 256 + 64 + d]  = tib;
    concat[bl * 256 + 128 + d] = f2bf(tc);
    qb[bl * 64 + d] = tib;
  }
  for (int i = tid; i < 8 * 256; i += THREADS) concat[8 * 256 + i] = 0;  // pad rows
  __syncthreads();

  // ---------------- Phase 1: attention MLP via WMMA ----------------
  unsigned short* h = hz + wave * 2048;            // per-wave [16][128] bf16
  const int mbase = (lane >= 16) ? 8 : 0;
#pragma unroll 1
  for (int mi = 0; mi < 4; ++mi) {
    int mt = wave * 4 + mi;                        // 16 row-tiles of (b,s) pairs
    int rr = mt * 16 + (lane & 15);
    const unsigned short* qrow = qb + (rr >> 5) * 64;
    const unsigned short* krow = keysb + ((rr >> 5) * 32 + (rr & 31)) * 64;

    v16bf afr[8];
#pragma unroll
    for (int kt = 0; kt < 8; ++kt) afr[kt] = build_feat_frag(qrow, krow, kt, lane);

    // layer 0: [16,256] @ aW0[256,128]
#pragma unroll 1
    for (int nt = 0; nt < 8; ++nt) {
      v8f acc = {};
#pragma unroll
      for (int kt = 0; kt < 8; ++kt)
        acc = wmma_bf16(afr[kt], load_b_frag(wsA0 + (kt * 8 + nt) * 256, lane), acc);
      int n = nt * 16 + (lane & 15);
      float bias = ab0[n];
#pragma unroll
      for (int g = 0; g < 8; ++g) {
        float v = acc[g] + bias;
        h[(mbase + g) * 128 + n] = f2bf(v > 0.f ? v : 0.f);
      }
    }

    // layer 1: [16,128] @ aW1[128,64]
    v16bf a1[4];
#pragma unroll
    for (int kt = 0; kt < 4; ++kt) a1[kt] = build_a_frag(h, 128, kt, lane);
    v8f acc1[4];
#pragma unroll
    for (int nt = 0; nt < 4; ++nt) {
      v8f acc = {};
#pragma unroll
      for (int kt = 0; kt < 4; ++kt)
        acc = wmma_bf16(a1[kt], load_b_frag(wsA1 + (kt * 4 + nt) * 256, lane), acc);
      acc1[nt] = acc;
    }

    // layer 2: bias+relu then dot with aW2 -> score; reduce across 16-lane half
#pragma unroll
    for (int g = 0; g < 8; ++g) {
      float p = 0.f;
#pragma unroll
      for (int nt = 0; nt < 4; ++nt) {
        int n = nt * 16 + (lane & 15);
        float v = acc1[nt][g] + ab1[n];
        p += (v > 0.f ? v : 0.f) * aW2[n];
      }
      p += __shfl_xor(p, 1, 32);
      p += __shfl_xor(p, 2, 32);
      p += __shfl_xor(p, 4, 32);
      p += __shfl_xor(p, 8, 32);
      if ((lane & 15) == 0) {
        int r = mt * 16 + mbase + g;
        attn[(r >> 5) * 32 + (r & 31)] = p + ab2[0];
      }
    }
  }
  __syncthreads();

  // ---------------- Phase 2: masked softmax ----------------
  if (tid < BT) {
    int len = seq_lengths[b0 + tid];
    len = len < 1 ? 1 : (len > S_SEQ ? S_SEQ : len);
    float* row = attn + tid * 32;
    float mx = -1e30f;
    for (int s = 0; s < len; ++s) mx = fmaxf(mx, row[s]);
    float sum = 0.f;
    for (int s = 0; s < SP; ++s) {
      float e = (s < len) ? __expf(row[s] - mx) : 0.f;
      row[s] = e;
      sum += e;
    }
    float inv = 1.f / sum;
    for (int s = 0; s < SP; ++s) row[s] *= inv;
  }
  __syncthreads();

  // attn output + interest = attn @ keys
  for (int i = tid; i < BT * S_SEQ; i += THREADS)
    attn_out[(size_t)(b0 + i / S_SEQ) * S_SEQ + (i % S_SEQ)] =
        attn[(i / S_SEQ) * 32 + (i % S_SEQ)];
  for (int i = tid; i < BT * 64; i += THREADS) {
    int bl = i >> 6, d = i & 63;
    const float* arow = attn + bl * 32;
    const unsigned short* kcol = keysb + bl * 32 * 64 + d;
    float sum = 0.f;
#pragma unroll 5
    for (int s = 0; s < S_SEQ; ++s) sum += arow[s] * bf2f(kcol[s * 64]);
    concat[bl * 256 + 192 + d] = f2bf(sum);
  }
  __syncthreads();

  // ---------------- Phase 3: final MLP via WMMA ----------------
  unsigned short* z  = hz;              // [16][256] bf16
  unsigned short* z2 = hz + 16 * 256;   // [16][128] bf16
  {
    v16bf a0[8];
#pragma unroll
    for (int kt = 0; kt < 8; ++kt) a0[kt] = build_a_frag(concat, 256, kt, lane);
#pragma unroll 1
    for (int ni = 0; ni < 4; ++ni) {    // wave handles 4 of 16 N-tiles
      int nt = wave * 4 + ni;
      v8f acc = {};
#pragma unroll
      for (int kt = 0; kt < 8; ++kt)
        acc = wmma_bf16(a0[kt], load_b_frag(wsM0 + (kt * 16 + nt) * 256, lane), acc);
      int n = nt * 16 + (lane & 15);
      float bias = mb0[n];
#pragma unroll
      for (int g = 0; g < 8; ++g) {
        float v = acc[g] + bias;
        z[(mbase + g) * 256 + n] = f2bf(v > 0.f ? v : 0.f);
      }
    }
  }
  __syncthreads();
  {
    v16bf a1[8];
#pragma unroll
    for (int kt = 0; kt < 8; ++kt) a1[kt] = build_a_frag(z, 256, kt, lane);
#pragma unroll 1
    for (int ni = 0; ni < 2; ++ni) {    // wave handles 2 of 8 N-tiles
      int nt = wave * 2 + ni;
      v8f acc = {};
#pragma unroll
      for (int kt = 0; kt < 8; ++kt)
        acc = wmma_bf16(a1[kt], load_b_frag(wsM1 + (kt * 8 + nt) * 256, lane), acc);
      int n = nt * 16 + (lane & 15);
      float bias = mb1[n];
#pragma unroll
      for (int g = 0; g < 8; ++g) {
        float v = acc[g] + bias;
        z2[(mbase + g) * 128 + n] = f2bf(v > 0.f ? v : 0.f);
      }
    }
  }
  __syncthreads();
  if (tid < BT) {
    const unsigned short* zrow = z2 + tid * 128;
    float s = mb2[0];
    for (int c = 0; c < 128; ++c) s += bf2f(zrow[c]) * mW2[c];
    out[b0 + tid] = 1.f / (1.f + __expf(-s));
  }
}

extern "C" void kernel_launch(void* const* d_in, const int* in_sizes, int n_in,
                              void* d_out, int out_size, void* d_ws, size_t ws_size,
                              hipStream_t stream) {
  (void)in_sizes; (void)n_in; (void)out_size; (void)ws_size;
  const int*   user_ids      = (const int*)d_in[0];
  const int*   target_items  = (const int*)d_in[1];
  const int*   target_cates  = (const int*)d_in[2];
  const int*   seq_items     = (const int*)d_in[3];
  const int*   seq_cates     = (const int*)d_in[4];
  const int*   seq_behaviors = (const int*)d_in[5];
  const int*   seq_lengths   = (const int*)d_in[6];
  const float* user_emb      = (const float*)d_in[7];
  const float* item_emb      = (const float*)d_in[8];
  const float* cate_emb      = (const float*)d_in[9];
  const float* behavior_emb  = (const float*)d_in[10];
  const float* aW0 = (const float*)d_in[11];
  const float* ab0 = (const float*)d_in[12];
  const float* aW1 = (const float*)d_in[13];
  const float* ab1 = (const float*)d_in[14];
  const float* aW2 = (const float*)d_in[15];
  const float* ab2 = (const float*)d_in[16];
  const float* mW0 = (const float*)d_in[17];
  const float* mb0 = (const float*)d_in[18];
  const float* mW1 = (const float*)d_in[19];
  const float* mb1 = (const float*)d_in[20];
  const float* mW2 = (const float*)d_in[21];
  const float* mb2 = (const float*)d_in[22];

  char* ws = (char*)d_ws;
  unsigned int* wsA0 = (unsigned int*)(ws);                      //  64 KB
  unsigned int* wsA1 = (unsigned int*)(ws + 65536);              //  16 KB
  unsigned int* wsM0 = (unsigned int*)(ws + 65536 + 16384);      // 128 KB
  unsigned int* wsM1 = (unsigned int*)(ws + 65536 + 16384 + 131072); // 64 KB

  DIN_swizzleW<<<64, 256, 0, stream>>>(aW0, wsA0, 256, 128);
  DIN_swizzleW<<<16, 256, 0, stream>>>(aW1, wsA1, 128, 64);
  DIN_swizzleW<<<128, 256, 0, stream>>>(mW0, wsM0, 256, 256);
  DIN_swizzleW<<<64, 256, 0, stream>>>(mW1, wsM1, 256, 128);

  float* out = (float*)d_out;
  float* attn_out = out + B_TOTAL;
  DIN_fused<<<B_TOTAL / BT, THREADS, 0, stream>>>(
      user_ids, target_items, target_cates, seq_items, seq_cates, seq_behaviors,
      seq_lengths, user_emb, item_emb, cate_emb, behavior_emb,
      ab0, ab1, aW2, ab2, mb0, mb1, mW2, mb2,
      wsA0, wsA1, wsM0, wsM1, out, attn_out);
}
